// MemoryLayerAttention_49701361549859
// MI455X (gfx1250) — matrix operand order repaired and checked by reading, not compiled, not run on hardware
//
#include <hip/hip_runtime.h>

typedef _Float16 h16 __attribute__((ext_vector_type(16)));
typedef _Float16 h8  __attribute__((ext_vector_type(8)));
typedef float    f8  __attribute__((ext_vector_type(8)));

#define WAVES 8

constexpr int DIM = 64, EMB = 24, HEADS = 4, FF = 64, MR = 16, MC = 16;
constexpr int SEQ = 17, LV = 256, LQ = 128, NB = 16;
constexpr int NSAMP = NB * LQ;           // 2048 samples
constexpr int FLATN = SEQ * EMB;         // 408

// swizzled B-tile table (each tile: 32 lanes x 16 halves = 1KB, fragment-ordered)
constexpr int TW_WIN   = 0;    // 8  : W_in_emb  [128,24]  (ks 0..3, nt 0..1)
constexpr int TW_WMEM  = 8;    // 2  : W_mem_emb [16,24]   (nt 0..1)
constexpr int TW_QKV   = 10;   // 24 : Wq/Wk/Wv  [24,96]   ((p*4+h)*2+nt)
constexpr int TW_WO    = 34;   // 8  : Wo        [96,24]   (h*2+nt)
constexpr int TW_FF1   = 42;   // 4  : W_ff1     [24,64]   (nt 0..3)
constexpr int TW_FF2   = 46;   // 4  : W_ff2     [64,24]   (ks*2+nt)
constexpr int TW_MEMIN = 50;   // 26 : W_memin   [408,32]  (ks*2+nt)
constexpr int NTILES   = 76;

// LDS constant pool offsets (floats)
constexpr int CB_IN  = 0;    // b_in_emb   24
constexpr int CB_MEM = 24;   // b_mem_emb  24
constexpr int CB_Q   = 48;   // bq         96
constexpr int CB_K   = 144;  // bk         96
constexpr int CB_V   = 240;  // bv         96
constexpr int CB_O   = 336;  // bo         24
constexpr int CB_F1  = 360;  // b_ff1      64
constexpr int CB_F2  = 424;  // b_ff2      24
constexpr int CB_LNG = 448;  // ln_g       24
constexpr int CB_LNB = 472;  // ln_b       24
constexpr int CB_MI  = 496;  // b_memin    32
constexpr int CB_OUT = 528;  // b_out      64
constexpr int NCST   = 592;

struct __attribute__((aligned(32))) SMem {
  _Float16 bt[NTILES][512];        // 76KB swizzled weight tiles
  float    cst[NCST];              // biases / LN params
  float    peL[SEQ * EMB];         // positional encoding
  float    miL[WAVES][32];         // memin results per sample
  float    einL[WAVES][32];        // emb_in results per sample
  _Float16 inb16[WAVES][128];      // [q||v_t] per sample, f16
  _Float16 flatL[WAVES][416];      // x2 flattened (408 + pad)
  _Float16 xL[WAVES][32][32];      // x / x1 (rows>=17, cols>=24 stay 0)
  _Float16 qL[WAVES][32][32];      // q_h then attn probs
  _Float16 kL[WAVES][32][32];
  _Float16 vB[WAVES][2][512];      // v_h pre-swizzled in B-fragment layout
  float    scr[WAVES][1024];       // scores f32 / ctx f16 / ffn hidden f16 / LN staging
};

__device__ inline f8 zero8() {
  f8 z;
#pragma unroll
  for (int i = 0; i < 8; ++i) z[i] = 0.f;
  return z;
}

__device__ inline h16 mk16(h8 lo, h8 hi) {
  h16 a;
#pragma unroll
  for (int e = 0; e < 8; ++e) { a[e] = lo[e]; a[8 + e] = hi[e]; }
  return a;
}

__device__ inline f8 wmma(h16 a, h16 b, f8 c) {
  return __builtin_amdgcn_wmma_f32_16x16x32_f16(false, a, false, b, (short)0, c, false, false);
}

// A fragment 16x32: lanes 0-15 -> M=lane, K {0..7,16..23}; lanes 16-31 -> K {8..15,24..31}
__device__ inline h16 load_a(const _Float16* p, int ld, int row0, int k0) {
  const int l = threadIdx.x & 31;
  const _Float16* r = p + (row0 + (l & 15)) * ld + k0 + ((l & 16) ? 8 : 0);
  return mk16(*(const h8*)r, *(const h8*)(r + 16));
}

// A fragment with M = sample-in-workgroup over a per-sample flat row (stride halves)
__device__ inline h16 load_a_batch(const _Float16* base, int stride, int k0) {
  const int l = threadIdx.x & 31;
  const int m = l & 15;
  h16 a;
  if (m < WAVES) {
    const _Float16* r = base + m * stride + k0 + ((l & 16) ? 8 : 0);
    a = mk16(*(const h8*)r, *(const h8*)(r + 16));
  } else {
#pragma unroll
    for (int e = 0; e < 16; ++e) a[e] = (_Float16)0.f;
  }
  return a;
}

// B fragment from a pre-swizzled tile: one contiguous 32B read per lane
__device__ inline h16 load_b_sw(const _Float16* tile) {
  return *(const h16*)(tile + (threadIdx.x & 31) * 16);
}

// B fragment where memory holds the transpose (k^T): two 16B reads per lane (K {0..7,16..23} / {8..15,24..31})
__device__ inline h16 load_bt(const _Float16* p, int ld, int k0, int n0) {
  const int l = threadIdx.x & 31;
  const _Float16* r = p + (n0 + (l & 15)) * ld + k0 + ((l & 16) ? 8 : 0);
  return mk16(*(const h8*)r, *(const h8*)(r + 16));
}

// B fragment straight from a global f32 matrix (W_out at the final step only)
__device__ inline h16 load_b_g(const float* g, int ld, int k0, int n0, int kmax) {
  const int l = threadIdx.x & 31;
  const int n = n0 + (l & 15);
  const int ko = (l & 16) ? 8 : 0;
  h16 b;
#pragma unroll
  for (int e = 0; e < 16; ++e) {
    int k = k0 + ko + e + ((e & 8) ? 8 : 0);
    b[e] = (k < kmax) ? (_Float16)g[k * ld + n] : (_Float16)0.f;
  }
  return b;
}

// stage one 32x16 B tile in fragment order (zero-padded outside [rows,cols))
__device__ inline void stage_tile(_Float16* dst, const float* src, int rows, int cols,
                                  int ld, int k0, int n0, int tid, int nthr) {
  for (int i = tid; i < 512; i += nthr) {
    int lane = i >> 4, e = i & 15;
    int k = k0 + ((lane & 16) ? 8 : 0) + e + ((e & 8) ? 8 : 0);
    int n = n0 + (lane & 15);
    dst[i] = (_Float16)((k < rows && n < cols) ? src[k * ld + n] : 0.f);
  }
}

__device__ inline void stage_vec(float* dst, const float* src, int n, int tid, int nthr) {
  for (int i = tid; i < n; i += nthr) dst[i] = src[i];
}

__global__ __launch_bounds__(WAVES * 32, 1)
void memcell_kernel(const float* __restrict__ queries, const float* __restrict__ values,
                    const float* __restrict__ W_in_emb, const float* __restrict__ b_in_emb,
                    const float* __restrict__ W_mem_emb, const float* __restrict__ b_mem_emb,
                    const float* __restrict__ Wq, const float* __restrict__ bq,
                    const float* __restrict__ Wk, const float* __restrict__ bk,
                    const float* __restrict__ Wv, const float* __restrict__ bv,
                    const float* __restrict__ Wo, const float* __restrict__ bo,
                    const float* __restrict__ W_ff1, const float* __restrict__ b_ff1,
                    const float* __restrict__ W_ff2, const float* __restrict__ b_ff2,
                    const float* __restrict__ ln_g, const float* __restrict__ ln_b,
                    const float* __restrict__ W_out, const float* __restrict__ b_out,
                    const float* __restrict__ W_memin, const float* __restrict__ b_memin,
                    float* __restrict__ out) {
  __shared__ SMem S;
  const int tid = threadIdx.x;
  const int nthr = blockDim.x;
  const int w = tid >> 5;
  const int l = tid & 31;
  const int samp = blockIdx.x * WAVES + w;
  const int bidx = samp / LQ;

  // ---------------- stage swizzled weight tiles + constants ----------------
#pragma unroll 1
  for (int i = 0; i < 8; ++i)
    stage_tile(S.bt[TW_WIN + i], W_in_emb, 128, EMB, EMB, (i >> 1) * 32, (i & 1) * 16, tid, nthr);
#pragma unroll 1
  for (int nt = 0; nt < 2; ++nt)
    stage_tile(S.bt[TW_WMEM + nt], W_mem_emb, MC, EMB, EMB, 0, nt * 16, tid, nthr);
#pragma unroll 1
  for (int i = 0; i < 24; ++i) {
    int p = i >> 3, h = (i >> 1) & 3, nt = i & 1;
    const float* Wp = (p == 0) ? Wq : (p == 1) ? Wk : Wv;
    stage_tile(S.bt[TW_QKV + i], Wp, EMB, 96, 96, 0, h * EMB + nt * 16, tid, nthr);
  }
#pragma unroll 1
  for (int i = 0; i < 8; ++i)
    stage_tile(S.bt[TW_WO + i], Wo, 96, EMB, EMB, (i >> 1) * EMB, (i & 1) * 16, tid, nthr);
#pragma unroll 1
  for (int nt = 0; nt < 4; ++nt)
    stage_tile(S.bt[TW_FF1 + nt], W_ff1, EMB, FF, FF, 0, nt * 16, tid, nthr);
#pragma unroll 1
  for (int i = 0; i < 4; ++i)
    stage_tile(S.bt[TW_FF2 + i], W_ff2, FF, EMB, EMB, (i >> 1) * 32, (i & 1) * 16, tid, nthr);
#pragma unroll 1
  for (int i = 0; i < 26; ++i)
    stage_tile(S.bt[TW_MEMIN + i], W_memin, FLATN, 32, 32, (i >> 1) * 32, (i & 1) * 16, tid, nthr);
  stage_vec(S.cst + CB_IN,  b_in_emb,  24, tid, nthr);
  stage_vec(S.cst + CB_MEM, b_mem_emb, 24, tid, nthr);
  stage_vec(S.cst + CB_Q,   bq,        96, tid, nthr);
  stage_vec(S.cst + CB_K,   bk,        96, tid, nthr);
  stage_vec(S.cst + CB_V,   bv,        96, tid, nthr);
  stage_vec(S.cst + CB_O,   bo,        24, tid, nthr);
  stage_vec(S.cst + CB_F1,  b_ff1,     64, tid, nthr);
  stage_vec(S.cst + CB_F2,  b_ff2,     24, tid, nthr);
  stage_vec(S.cst + CB_LNG, ln_g,      24, tid, nthr);
  stage_vec(S.cst + CB_LNB, ln_b,      24, tid, nthr);
  stage_vec(S.cst + CB_MI,  b_memin,   32, tid, nthr);
  stage_vec(S.cst + CB_OUT, b_out,     64, tid, nthr);
  // positional encoding
  for (int i = tid; i < SEQ * EMB; i += nthr) {
    int pos = i / EMB, d = i % EMB;
    float ang = (float)pos / powf(10000.f, (2.f * (float)(d / 2)) / (float)EMB);
    S.peL[i] = (d & 1) ? cosf(ang) : sinf(ang);
  }
  // per-wave init
#pragma unroll
  for (int c = 0; c < 32; ++c) {
    S.xL[w][l][c] = (_Float16)0.f;
    S.qL[w][l][c] = (_Float16)0.f;
    S.kL[w][l][c] = (_Float16)0.f;
  }
  if (l < 8) S.flatL[w][408 + l] = (_Float16)0.f;
  S.inb16[w][l]      = (_Float16)queries[samp * DIM + l];
  S.inb16[w][32 + l] = (_Float16)queries[samp * DIM + 32 + l];
  __syncthreads();

  _Float16* xw  = &S.xL[w][0][0];
  _Float16* qw  = &S.qL[w][0][0];
  _Float16* kw  = &S.kL[w][0][0];
  _Float16* vbw = &S.vB[w][0][0];
  float*    scw   = S.scr[w];
  _Float16* scw16 = reinterpret_cast<_Float16*>(scw);
  const float scale = 0.20412414523193154f;  // 1/sqrt(24)
  const int hi8 = (l & 16) ? 8 : 0;

  // memory state in registers, A-fragment layout: lane holds row (l&15), cols hi8..hi8+7
  float memr[8];
#pragma unroll
  for (int i = 0; i < 8; ++i) memr[i] = 1e-6f;

#pragma unroll 1
  for (int t = 0; t < LV; ++t) {
    // -------- stage this step's value row (f16) --------
    const float* vrow = values + (bidx * LV + t) * DIM;
    S.inb16[w][64 + l] = (_Float16)vrow[l];
    S.inb16[w][96 + l] = (_Float16)vrow[32 + l];
    __syncthreads();  // b1: inb16 ready

    // -------- memory embedding: mem[16,16] @ W_mem_emb -> x rows 1..16 (per wave) --------
    {
      h16 a;
#pragma unroll
      for (int e = 0; e < 8; ++e) a[e] = (_Float16)memr[e];
#pragma unroll
      for (int e = 0; e < 8; ++e) a[8 + e] = (_Float16)0.f;   // K=16..31 pad
#pragma unroll
      for (int nt = 0; nt < 2; ++nt) {
        h16 b = load_b_sw(S.bt[TW_WMEM + nt]);
        f8 c = wmma(a, b, zero8());
        int col = nt * 16 + (l & 15);
        if (col < EMB) {
#pragma unroll
          for (int j = 0; j < 8; ++j) {
            int row = 1 + j + hi8;
            xw[row * 32 + col] = (_Float16)(c[j] + S.cst[CB_MEM + col] + S.peL[row * EMB + col]);
          }
        }
      }
    }
    // -------- token-0 embedding batched over samples: [8,128]@[128,24] (waves 6,7) --------
    if (w >= 6) {
      int nt = w - 6;
      f8 c = zero8();
#pragma unroll
      for (int ks = 0; ks < 4; ++ks) {
        h16 a = load_a_batch(&S.inb16[0][0], 128, ks * 32);
        h16 b = load_b_sw(S.bt[TW_WIN + ks * 2 + nt]);
        c = wmma(a, b, c);
      }
      int col = nt * 16 + (l & 15);
      if (col < EMB) {
#pragma unroll
        for (int j = 0; j < 8; ++j) {
          int sm = j + hi8;
          if (sm < WAVES) S.einL[sm][col] = c[j] + S.cst[CB_IN + col];
        }
      }
    }
    __syncthreads();  // b2: einL ready
    if (l < EMB) xw[l] = (_Float16)(S.einL[w][l] + S.peL[l]);

    // -------- multi-head attention --------
    f8 oc[2][2];
#pragma unroll
    for (int i = 0; i < 2; ++i)
#pragma unroll
      for (int jn = 0; jn < 2; ++jn) oc[i][jn] = zero8();

#pragma unroll 1
    for (int h = 0; h < HEADS; ++h) {
      h16 xa0 = load_a(xw, 32, 0, 0);
      h16 xa1 = load_a(xw, 32, 16, 0);
      // q,k -> row-major; v -> B-fragment layout
#pragma unroll
      for (int p = 0; p < 3; ++p) {
        const float* bias = &S.cst[CB_Q + p * 96 + h * EMB];
#pragma unroll
        for (int nt = 0; nt < 2; ++nt) {
          h16 b = load_b_sw(S.bt[TW_QKV + (p * 4 + h) * 2 + nt]);
          f8 c0 = wmma(xa0, b, zero8());
          f8 c1 = wmma(xa1, b, zero8());
          int col = nt * 16 + (l & 15);
          float bb = (col < EMB) ? bias[col] : 0.f;
          if (p < 2) {
            _Float16* dst = (p == 0) ? qw : kw;
            if (col < EMB) {
#pragma unroll
              for (int j = 0; j < 8; ++j) {
                int m0 = j + hi8;
                if (m0 < SEQ)      dst[m0 * 32 + col]        = (_Float16)(c0[j] + bb);
                if (16 + m0 < SEQ) dst[(16 + m0) * 32 + col] = (_Float16)(c1[j] + bb);
              }
            }
          } else {
            h8 pk0, pk1;
#pragma unroll
            for (int j = 0; j < 8; ++j) {
              pk0[j] = (col < EMB) ? (_Float16)(c0[j] + bb) : (_Float16)0.f;
              pk1[j] = (col < EMB && (16 + j + hi8) < SEQ) ? (_Float16)(c1[j] + bb) : (_Float16)0.f;
            }
            _Float16* vt = vbw + nt * 512;
            *(h8*)(vt + l * 16)     = pk0;  // tokens hi8..hi8+7
            *(h8*)(vt + l * 16 + 8) = pk1;  // tokens 16+hi8..16+hi8+7
          }
        }
      }
      // scores = q @ k^T * scale -> scr (f32)
      {
        h16 qa0 = load_a(qw, 32, 0, 0);
        h16 qa1 = load_a(qw, 32, 16, 0);
#pragma unroll
        for (int nt = 0; nt < 2; ++nt) {
          h16 bt = load_bt(kw, 32, 0, nt * 16);
          f8 c0 = wmma(qa0, bt, zero8());
          f8 c1 = wmma(qa1, bt, zero8());
          int tok = nt * 16 + (l & 15);
          if (tok < SEQ) {
#pragma unroll
            for (int j = 0; j < 8; ++j) {
              int m0 = j + hi8;
              if (m0 < SEQ)      scw[m0 * 32 + tok]        = c0[j] * scale;
              if (16 + m0 < SEQ) scw[(16 + m0) * 32 + tok] = c1[j] * scale;
            }
          }
        }
      }
      // softmax over 17 keys (lane = row), probs into qL (q is dead)
      if (l < SEQ) {
        float mx = -1e30f;
        for (int t2 = 0; t2 < SEQ; ++t2) mx = fmaxf(mx, scw[l * 32 + t2]);
        float sum = 0.f;
        float pr[SEQ];
        for (int t2 = 0; t2 < SEQ; ++t2) { pr[t2] = __expf(scw[l * 32 + t2] - mx); sum += pr[t2]; }
        float inv = 1.f / sum;
        for (int t2 = 0; t2 < SEQ; ++t2) qw[l * 32 + t2] = (_Float16)(pr[t2] * inv);
        for (int t2 = SEQ; t2 < EMB; ++t2) qw[l * 32 + t2] = (_Float16)0.f;
      }
      // ctx = probs @ v -> scratch f16 (K-pad cols 24..31 re-zeroed, one 16B store)
      {
        h16 pa0 = load_a(qw, 32, 0, 0);
        h16 pa1 = load_a(qw, 32, 16, 0);
        h8 z8;
#pragma unroll
        for (int j = 0; j < 8; ++j) z8[j] = (_Float16)0.f;
        *(h8*)(scw16 + l * 32 + 24) = z8;
#pragma unroll
        for (int nt = 0; nt < 2; ++nt) {
          h16 b = load_b_sw(vbw + nt * 512);
          f8 c0 = wmma(pa0, b, zero8());
          f8 c1 = wmma(pa1, b, zero8());
          int col = nt * 16 + (l & 15);
          if (col < EMB) {
#pragma unroll
            for (int j = 0; j < 8; ++j) {
              int m0 = j + hi8;
              if (m0 < SEQ)      scw16[m0 * 32 + col]        = (_Float16)c0[j];
              if (16 + m0 < SEQ) scw16[(16 + m0) * 32 + col] = (_Float16)c1[j];
            }
          }
        }
      }
      // o += ctx @ Wo_h
      {
        h16 ca0 = load_a(scw16, 32, 0, 0);
        h16 ca1 = load_a(scw16, 32, 16, 0);
#pragma unroll
        for (int nt = 0; nt < 2; ++nt) {
          h16 b = load_b_sw(S.bt[TW_WO + h * 2 + nt]);
          oc[0][nt] = wmma(ca0, b, oc[0][nt]);
          oc[1][nt] = wmma(ca1, b, oc[1][nt]);
        }
      }
    }  // heads
    // residual + LN -> x1 (back into xw)
#pragma unroll
    for (int mt = 0; mt < 2; ++mt)
#pragma unroll
      for (int nt = 0; nt < 2; ++nt) {
        int col = nt * 16 + (l & 15);
        if (col < EMB) {
#pragma unroll
          for (int j = 0; j < 8; ++j) {
            int m = mt * 16 + j + hi8;
            if (m < SEQ)
              scw[m * 32 + col] = oc[mt][nt][j] + S.cst[CB_O + col] + (float)xw[m * 32 + col];
          }
        }
      }
    if (l < SEQ) {
      float mu = 0.f;
      for (int e = 0; e < EMB; ++e) mu += scw[l * 32 + e];
      mu *= (1.f / EMB);
      float var = 0.f;
      for (int e = 0; e < EMB; ++e) { float d = scw[l * 32 + e] - mu; var += d * d; }
      float rs = rsqrtf(var * (1.f / EMB) + 1e-6f);
      for (int e = 0; e < EMB; ++e)
        xw[l * 32 + e] =
            (_Float16)((scw[l * 32 + e] - mu) * rs * S.cst[CB_LNG + e] + S.cst[CB_LNB + e]);
    }
    // -------- FFN --------
    {
      h16 xa0 = load_a(xw, 32, 0, 0);
      h16 xa1 = load_a(xw, 32, 16, 0);
      f8 hc[2][4];
#pragma unroll
      for (int nt = 0; nt < 4; ++nt) {
        h16 b = load_b_sw(S.bt[TW_FF1 + nt]);
        hc[0][nt] = wmma(xa0, b, zero8());
        hc[1][nt] = wmma(xa1, b, zero8());
      }
#pragma unroll
      for (int mt = 0; mt < 2; ++mt)
#pragma unroll
        for (int nt = 0; nt < 4; ++nt) {
          int col = nt * 16 + (l & 15);
#pragma unroll
          for (int j = 0; j < 8; ++j) {
            int m = mt * 16 + j + hi8;
            if (m < SEQ)
              scw16[m * 64 + col] = (_Float16)fmaxf(hc[mt][nt][j] + S.cst[CB_F1 + col], 0.f);
          }
        }
      h16 ha[2][2];
      ha[0][0] = load_a(scw16, 64, 0, 0);
      ha[0][1] = load_a(scw16, 64, 0, 32);
      ha[1][0] = load_a(scw16, 64, 16, 0);
      ha[1][1] = load_a(scw16, 64, 16, 32);
      f8 fc[2][2];
#pragma unroll
      for (int i = 0; i < 2; ++i)
#pragma unroll
        for (int jn = 0; jn < 2; ++jn) fc[i][jn] = zero8();
#pragma unroll
      for (int ks = 0; ks < 2; ++ks)
#pragma unroll
        for (int nt = 0; nt < 2; ++nt) {
          h16 b = load_b_sw(S.bt[TW_FF2 + ks * 2 + nt]);
          fc[0][nt] = wmma(ha[0][ks], b, fc[0][nt]);
          fc[1][nt] = wmma(ha[1][ks], b, fc[1][nt]);
        }
#pragma unroll
      for (int mt = 0; mt < 2; ++mt)
#pragma unroll
        for (int nt = 0; nt < 2; ++nt) {
          int col = nt * 16 + (l & 15);
          if (col < EMB) {
#pragma unroll
            for (int j = 0; j < 8; ++j) {
              int m = mt * 16 + j + hi8;
              if (m < SEQ)
                scw[m * 32 + col] = fc[mt][nt][j] + S.cst[CB_F2 + col] + (float)xw[m * 32 + col];
            }
          }
        }
      // LN -> x2 straight into the flat buffer (only consumer is the flat GEMMs)
      if (l < SEQ) {
        float mu = 0.f;
        for (int e = 0; e < EMB; ++e) mu += scw[l * 32 + e];
        mu *= (1.f / EMB);
        float var = 0.f;
        for (int e = 0; e < EMB; ++e) { float d = scw[l * 32 + e] - mu; var += d * d; }
        float rs = rsqrtf(var * (1.f / EMB) + 1e-6f);
        for (int e = 0; e < EMB; ++e)
          S.flatL[w][l * EMB + e] =
              (_Float16)((scw[l * 32 + e] - mu) * rs * S.cst[CB_LNG + e] + S.cst[CB_LNB + e]);
      }
    }
    // -------- memin: FLAT[8,408] @ W_memin[408,32], batched over the workgroup --------
    __syncthreads();  // b3: flatL ready
    if (w < 2) {
      f8 c = zero8();
#pragma unroll 1
      for (int ks = 0; ks < 13; ++ks) {
        h16 a = load_a_batch(&S.flatL[0][0], 416, ks * 32);
        h16 b = load_b_sw(S.bt[TW_MEMIN + ks * 2 + w]);
        c = wmma(a, b, c);
      }
      int col = w * 16 + (l & 15);
#pragma unroll
      for (int j = 0; j < 8; ++j) {
        int sm = j + hi8;
        if (sm < WAVES) S.miL[sm][col] = c[j] + S.cst[CB_MI + col];
      }
    }
    __syncthreads();  // b4: miL ready
    // gated memory update (registers; lane = row l&15, cols hi8..hi8+7)
    {
      float cs = S.miL[w][l & 15];
      float sp = 1.f / (1.f + __expf(-cs));  // sigmoid(c)
      float sn = 1.f - sp;                   // sigmoid(-c)
#pragma unroll
      for (int i = 0; i < 8; ++i)
        memr[i] = sn * memr[i] + sp * S.miL[w][MR + hi8 + i];
    }
    // -------- final output projection, only at the last step --------
    if (t == LV - 1 && w < 4) {
      f8 c = zero8();
#pragma unroll 1
      for (int ks = 0; ks < 13; ++ks) {
        h16 a = load_a_batch(&S.flatL[0][0], 416, ks * 32);
        h16 b = load_b_g(W_out, DIM, ks * 32, w * 16, FLATN);
        c = wmma(a, b, c);
      }
      int col = w * 16 + (l & 15);
#pragma unroll
      for (int j = 0; j < 8; ++j) {
        int sm = j + hi8;
        if (sm < WAVES) out[(blockIdx.x * WAVES + sm) * DIM + col] = c[j] + S.cst[CB_OUT + col];
      }
    }
  }  // t
}

extern "C" void kernel_launch(void* const* d_in, const int* in_sizes, int n_in,
                              void* d_out, int out_size, void* d_ws, size_t ws_size,
                              hipStream_t stream) {
  (void)in_sizes; (void)n_in; (void)out_size; (void)d_ws; (void)ws_size;
  const float* A[24];
  for (int i = 0; i < 24; ++i) A[i] = (const float*)d_in[i];
  dim3 grid(NSAMP / WAVES);   // 256 workgroups
  dim3 block(WAVES * 32);     // 8 waves = 256 threads
  memcell_kernel<<<grid, block, 0, stream>>>(
      A[0], A[1], A[2], A[3], A[4], A[5], A[6], A[7], A[8], A[9], A[10], A[11],
      A[12], A[13], A[14], A[15], A[16], A[17], A[18], A[19], A[20], A[21], A[22], A[23],
      (float*)d_out);
}
